// LLMFactFormer2D_6760278523966
// MI455X (gfx1250) — matrix operand order repaired and checked by reading, not compiled, not run on hardware
//
#include <hip/hip_runtime.h>
#include <hip/hip_bf16.h>
#include <math.h>

// ---------------------------------------------------------------------------
// CDNA5 (gfx1250) WMMA types
// ---------------------------------------------------------------------------
typedef __attribute__((ext_vector_type(16))) _Float16 v16h;
typedef __attribute__((ext_vector_type(8)))  float    v8f;
typedef __attribute__((ext_vector_type(4)))  _Float16 h4;
typedef float float4u __attribute__((ext_vector_type(4), aligned(4)));

union HV16 { v16h v; _Float16 h[16]; };
union FV8  { v8f  v; float    f[8];  };

__device__ __forceinline__ float gelu_f(float x) {
  float x3 = x * x * x;
  return 0.5f * x * (1.f + tanhf(0.79788456080286536f * (x + 0.044715f * x3)));
}
template<int ACT> __device__ __forceinline__ float act_f(float x) {
  if (ACT == 1) return gelu_f(x);
  if (ACT == 2) return fmaxf(x, 0.f);
  return x;
}

// ---------------------------------------------------------------------------
// Wave-level 16x16x32 f16 WMMA on a 64x64 LDS tile (wave -> 16 rows x 32 cols)
// ---------------------------------------------------------------------------
__device__ __forceinline__ void wmma_tile(const _Float16 (&As)[64][40],
                                          const _Float16 (&Ws)[64][40],
                                          int wm, int wn, int lane,
                                          FV8& acc0, FV8& acc1) {
  HV16 a, b0, b1;
  int am = wm * 16 + (lane & 15);
  int kg = (lane >> 4) * 16;         // A: lanes0-15 -> K 0..15, lanes16-31 -> K 16..31
#pragma unroll
  for (int e = 0; e < 16; ++e) { a.h[e] = As[am][kg + e]; }
  int bn = wn * 32 + (lane & 15);
  int kb = (lane >> 4) * 16;         // B: lanes0-15 -> K 0..15, lanes16-31 -> K 16..31
#pragma unroll
  for (int e = 0; e < 16; ++e) { b0.h[e] = Ws[bn][kb + e]; b1.h[e] = Ws[bn + 16][kb + e]; }
  acc0.v = __builtin_amdgcn_wmma_f32_16x16x32_f16(false, a.v, false, b0.v, (short)0, acc0.v, false, false);
  acc1.v = __builtin_amdgcn_wmma_f32_16x16x32_f16(false, a.v, false, b1.v, (short)0, acc1.v, false, false);
}

// fast (full-tile) loader: 64 rows x 32 cols of f32 at src(ld) -> f16 LDS tile
__device__ __forceinline__ void load_tile_fast(const float* __restrict__ src, long ld,
                                               _Float16 (&dst)[64][40], int tid) {
#pragma unroll
  for (int g = 0; g < 2; ++g) {
    int r = (tid >> 3) + g * 32;
    int c = (tid & 7) * 4;
    float4u v = *(const float4u*)(src + (long)r * ld + c);
    h4 hv;
    hv[0] = (_Float16)v[0]; hv[1] = (_Float16)v[1];
    hv[2] = (_Float16)v[2]; hv[3] = (_Float16)v[3];
    *(h4*)(&dst[r][c]) = hv;
  }
}

// ---------------------------------------------------------------------------
// Generic batched GEMM: C[z] = act(A[z] (MxK) * B[z] + bias) (+res)
// BT=true : B is [N,K] row-major (weights, "B^T")
// BT=false: B is [K,N] row-major
// batch offsets: off = (z/zdiv)*zhi + (z%zdiv)*zlo
// store index:   idx = offC + (m/mdiv)*pMhi + (m%mdiv)*pMlo + (n/ndiv)*pNhi + n%ndiv
// ---------------------------------------------------------------------------
template<int ACT, bool BT>
__global__ __launch_bounds__(256) void gemm_wmma(
    const float* __restrict__ A, const float* __restrict__ B,
    const float* __restrict__ bias, const float* __restrict__ res,
    float* __restrict__ C,
    int M, int N, int K, int lda, int ldb,
    int zdA, long zhA, long zlA,
    int zdB, long zhB, long zlB,
    int zdC, long zhC, long zlC,
    int mdiv, long pMhi, long pMlo, int ndiv, long pNhi) {
  __shared__ _Float16 As[64][40];
  __shared__ _Float16 Ws[64][40];
  int z = blockIdx.z;
  long offA = (long)(z / zdA) * zhA + (long)(z % zdA) * zlA;
  long offB = (long)(z / zdB) * zhB + (long)(z % zdB) * zlB;
  long offC = (long)(z / zdC) * zhC + (long)(z % zdC) * zlC;
  int m0 = blockIdx.y * 64, n0 = blockIdx.x * 64;
  int tid = threadIdx.x, lane = tid & 31, wave = tid >> 5;
  int wm = wave >> 1, wn = wave & 1;
  bool fullM = (m0 + 64 <= M);
  bool fullN = (n0 + 64 <= N);
  FV8 acc0, acc1;
#pragma unroll
  for (int i = 0; i < 8; ++i) { acc0.f[i] = 0.f; acc1.f[i] = 0.f; }

  for (int k0 = 0; k0 < K; k0 += 32) {
    bool fullK = (k0 + 32 <= K);
    __syncthreads();
    // ---- A tile
    if (fullM && fullK) {
      load_tile_fast(A + offA + (long)m0 * lda + k0, lda, As, tid);
    } else {
#pragma unroll
      for (int t = 0; t < 8; ++t) {
        int idx = tid + t * 256;
        int r = idx >> 5, c = idx & 31;
        int gm = m0 + r, gk = k0 + c;
        float va = 0.f;
        if (gm < M && gk < K) va = A[offA + (long)gm * lda + gk];
        As[r][c] = (_Float16)va;
      }
    }
    // ---- B tile (stored as [n][k] in LDS)
    if (BT) {
      if (fullN && fullK) {
        load_tile_fast(B + offB + (long)n0 * ldb + k0, ldb, Ws, tid);
      } else {
#pragma unroll
        for (int t = 0; t < 8; ++t) {
          int idx = tid + t * 256;
          int r = idx >> 5, c = idx & 31;
          int gn = n0 + r, gk = k0 + c;
          float vb = 0.f;
          if (gn < N && gk < K) vb = B[offB + (long)gn * ldb + gk];
          Ws[r][c] = (_Float16)vb;
        }
      }
    } else {
      if (fullN && fullK) {
#pragma unroll
        for (int t = 0; t < 8; ++t) {
          int idx = tid + t * 256;
          int r = idx & 63, c = idx >> 6;   // consecutive lanes -> consecutive n (coalesced)
          Ws[r][c] = (_Float16)B[offB + (long)(k0 + c) * ldb + (n0 + r)];
        }
      } else {
#pragma unroll
        for (int t = 0; t < 8; ++t) {
          int idx = tid + t * 256;
          int r = idx & 63, c = idx >> 6;
          int gn = n0 + r, gk = k0 + c;
          float vb = 0.f;
          if (gn < N && gk < K) vb = B[offB + (long)gk * ldb + gn];
          Ws[r][c] = (_Float16)vb;
        }
      }
    }
    __syncthreads();
    wmma_tile(As, Ws, wm, wn, lane, acc0, acc1);
  }

#pragma unroll
  for (int sub = 0; sub < 2; ++sub) {
    FV8 acc = sub ? acc1 : acc0;
#pragma unroll
    for (int r = 0; r < 8; ++r) {
      int m = m0 + wm * 16 + r + 8 * (lane >> 4);
      int n = n0 + wn * 32 + sub * 16 + (lane & 15);
      if (m < M && n < N) {
        float v = acc.f[r];
        if (bias) v += bias[n];
        v = act_f<ACT>(v);
        long idx = offC + (long)(m / mdiv) * pMhi + (long)(m % mdiv) * pMlo
                        + (long)(n / ndiv) * pNhi + (long)(n % ndiv);
        if (res) v += res[idx];
        C[idx] = v;
      }
    }
  }
}

// ---------------------------------------------------------------------------
// im2col conv GEMM: M=900 (b,oh,ow), N=256 out-channels, K=65536=(i,kh,kw)
// A gathered from token tensor U(b,nx,ny,256); fused bias + GELU.
// ---------------------------------------------------------------------------
__global__ __launch_bounds__(256) void conv_gemm_kernel(
    const float* __restrict__ U, const float* __restrict__ W,
    const float* __restrict__ bias, float* __restrict__ C) {
  __shared__ _Float16 As[64][40];
  __shared__ _Float16 Ws[64][40];
  const int M = 900, N = 256, K = 65536;
  int m0 = blockIdx.y * 64, n0 = blockIdx.x * 64;
  int tid = threadIdx.x, lane = tid & 31, wave = tid >> 5;
  int wm = wave >> 1, wn = wave & 1;
  bool fullM = (m0 + 64 <= M);
  FV8 acc0, acc1;
#pragma unroll
  for (int i = 0; i < 8; ++i) { acc0.f[i] = 0.f; acc1.f[i] = 0.f; }

  for (int k0 = 0; k0 < K; k0 += 32) {
    __syncthreads();
    // gathered A tile (im2col)
    if (fullM) {
#pragma unroll
      for (int t = 0; t < 8; ++t) {
        int idx = tid + t * 256;
        int r = idx >> 5, c = idx & 31;
        int gm = m0 + r, gk = k0 + c;
        int b = gm / 225, rr = gm % 225, oh = rr / 15, ow = rr % 15;
        int i = gk >> 8, kh = (gk >> 4) & 15, kw = gk & 15;
        As[r][c] = (_Float16)U[(((long)(b * 128 + oh * 8 + kh)) * 128 + (ow * 8 + kw)) * 256 + i];
      }
    } else {
#pragma unroll
      for (int t = 0; t < 8; ++t) {
        int idx = tid + t * 256;
        int r = idx >> 5, c = idx & 31;
        int gm = m0 + r, gk = k0 + c;
        float va = 0.f;
        if (gm < M) {
          int b = gm / 225, rr = gm % 225, oh = rr / 15, ow = rr % 15;
          int i = gk >> 8, kh = (gk >> 4) & 15, kw = gk & 15;
          va = U[(((long)(b * 128 + oh * 8 + kh)) * 128 + (ow * 8 + kw)) * 256 + i];
        }
        As[r][c] = (_Float16)va;
      }
    }
    load_tile_fast(W + (long)n0 * 65536 + k0, 65536, Ws, tid);
    __syncthreads();
    wmma_tile(As, Ws, wm, wn, lane, acc0, acc1);
  }
#pragma unroll
  for (int sub = 0; sub < 2; ++sub) {
    FV8 acc = sub ? acc1 : acc0;
#pragma unroll
    for (int r = 0; r < 8; ++r) {
      int m = m0 + wm * 16 + r + 8 * (lane >> 4);
      int n = n0 + wn * 32 + sub * 16 + (lane & 15);
      if (m < M && n < N) C[(long)m * 256 + n] = gelu_f(acc.f[r] + bias[n]);
    }
  }
}

// ---------------------------------------------------------------------------
// Elementwise / normalization helpers
// ---------------------------------------------------------------------------
__global__ __launch_bounds__(256) void ln_kernel(const float* __restrict__ X,
    const float* __restrict__ G, const float* __restrict__ Bb,
    float* __restrict__ Y, int R, int D, float eps) {
  int wave = threadIdx.x >> 5, lane = threadIdx.x & 31;
  int row = blockIdx.x * 8 + wave;
  if (row >= R) return;
  const float* x = X + (long)row * D;
  float s = 0.f;
  for (int i = lane; i < D; i += 32) s += x[i];
  for (int o = 16; o > 0; o >>= 1) s += __shfl_xor(s, o, 32);
  float mean = s / D;
  float v = 0.f;
  for (int i = lane; i < D; i += 32) { float d = x[i] - mean; v += d * d; }
  for (int o = 16; o > 0; o >>= 1) v += __shfl_xor(v, o, 32);
  float rstd = rsqrtf(v / D + eps);
  float* y = Y + (long)row * D;
  for (int i = lane; i < D; i += 32) y[i] = (x[i] - mean) * rstd * G[i] + Bb[i];
}

__global__ __launch_bounds__(256) void gn_stats_kernel(const float* __restrict__ X,
    float* __restrict__ stats, int zdiv, long zhi, long zlo,
    int T, int Cc, long strideT, float eps) {
  int z = blockIdx.x;
  long base = (long)(z / zdiv) * zhi + (long)(z % zdiv) * zlo;
  long n = (long)T * Cc;
  float s = 0.f, s2 = 0.f;
  for (long i = threadIdx.x; i < n; i += 256) {
    long t = i / Cc; int c = (int)(i % Cc);
    float v = X[base + t * strideT + c];
    s += v; s2 += v * v;
  }
  __shared__ float sh[256], sh2[256];
  sh[threadIdx.x] = s; sh2[threadIdx.x] = s2;
  __syncthreads();
  for (int o = 128; o > 0; o >>= 1) {
    if (threadIdx.x < o) { sh[threadIdx.x] += sh[threadIdx.x + o]; sh2[threadIdx.x] += sh2[threadIdx.x + o]; }
    __syncthreads();
  }
  if (threadIdx.x == 0) {
    float m = sh[0] / (float)n;
    float var = sh2[0] / (float)n - m * m;
    stats[z * 2] = m;
    stats[z * 2 + 1] = rsqrtf(var + eps);
  }
}

// attention GN apply: O2(h,i,l,c) per batch -> OG tokens (i*128+l, h*64+c)
__global__ void og_apply_attn_kernel(const float* __restrict__ O2,
    const float* __restrict__ stats, const float* __restrict__ G,
    const float* __restrict__ Bb, float* __restrict__ OG) {
  long idx = (long)blockIdx.x * blockDim.x + threadIdx.x;
  if (idx >= 8388608L) return;
  int h = (int)(idx >> 20);
  long rem = idx & 1048575L;
  long t = rem >> 6; int c = (int)(rem & 63);
  int ch = h * 64 + c;
  float v = (O2[idx] - stats[h * 2]) * stats[h * 2 + 1] * G[ch] + Bb[ch];
  OG[t * 512 + ch] = v;
}

// final GN apply on token-major (b,t,256), groups of 32 channels
__global__ void og_apply_final_kernel(const float* __restrict__ X,
    const float* __restrict__ stats, const float* __restrict__ G,
    const float* __restrict__ Bb, float* __restrict__ Y) {
  long idx = (long)blockIdx.x * blockDim.x + threadIdx.x;
  if (idx >= 16777216L) return;
  int b = (int)(idx >> 22);
  int ch = (int)(idx & 255);
  int z = b * 8 + (ch >> 5);
  Y[idx] = (X[idx] - stats[z * 2]) * stats[z * 2 + 1] * G[ch] + Bb[ch];
}

__global__ void add_bcast_kernel(float* __restrict__ X, const float* __restrict__ V,
                                 long n, long mod) {
  long idx = (long)blockIdx.x * blockDim.x + threadIdx.x;
  if (idx < n) X[idx] += V[idx % mod];
}

__global__ void pool_mean_kernel(const float* __restrict__ in, float* __restrict__ out,
                                 int zdiv, long zhi, long zlo, long strideJ, int J) {
  int z = blockIdx.x, c = threadIdx.x;
  long base = (long)(z / zdiv) * zhi + (long)(z % zdiv) * zlo;
  float s = 0.f;
  for (int j = 0; j < J; ++j) s += in[base + (long)j * strideJ + c];
  out[(long)z * 256 + c] = s * (1.f / (float)J);
}

__global__ void extract_pos_kernel(const float* __restrict__ grid, float* __restrict__ POS) {
  int i = threadIdx.x;
  if (i < 128) {
    POS[i]       = grid[(long)i * 256];   // grid[0,i,0,0]
    POS[128 + i] = grid[(long)i * 2 + 1]; // grid[0,0,i,1]
  }
}

__global__ void pe_feat_kernel(const float* __restrict__ POSX, const float* __restrict__ POSY,
                               const float* __restrict__ Bg, float* __restrict__ S) {
  long idx = (long)blockIdx.x * blockDim.x + threadIdx.x;
  if (idx >= 2097152L) return; // 16384*128
  int f = (int)(idx & 127);
  long t = idx >> 7;
  int i = (int)(t >> 7), j = (int)(t & 127);
  float ff = 6.283185307179586f * (POSX[i] * Bg[f] + POSY[j] * Bg[128 + f]);
  S[t * 256 + f] = sinf(ff);
  S[t * 256 + 128 + f] = cosf(ff);
}

// RoPE + head split: QK rows (b*128+n, 2048) -> QR/KR (b,h,n,d)
__global__ void rope_kernel(const float* __restrict__ QK, const float* __restrict__ POS,
                            float* __restrict__ QR, float* __restrict__ KR) {
  long idx = (long)blockIdx.x * blockDim.x + threadIdx.x;
  if (idx >= 524288L) return;
  int d = (int)(idx & 127);
  int n = (int)((idx >> 7) & 127);
  int h = (int)((idx >> 14) & 7);
  int b = (int)(idx >> 17);
  long base = ((long)(b * 128 + n)) * 2048 + h * 128;
  int d2 = d & 63;
  float inv = powf(10000.f, -(float)(2 * d2) * (1.f / 128.f));
  float f = POS[n] * 64.f * inv;
  float cs = cosf(f), sn = sinf(f);
  long outi = (((long)(b * 8 + h)) * 128 + n) * 128 + d;
  float q  = QK[base + d];
  float qr = (d < 64) ? -QK[base + d + 64] : QK[base + d - 64];
  QR[outi] = q * cs + qr * sn;
  float k  = QK[base + 1024 + d];
  float kr = (d < 64) ? -QK[base + 1024 + d + 64] : QK[base + 1024 + d - 64];
  KR[outi] = k * cs + kr * sn;
}

// (b,225,256) token-major -> (b,256,225)
__global__ void transpose_conv_kernel(const float* __restrict__ in, float* __restrict__ out) {
  long idx = (long)blockIdx.x * blockDim.x + threadIdx.x;
  if (idx >= 230400L) return;
  int b = (int)(idx / 57600);
  int rem = (int)(idx % 57600);
  int p = rem / 256, o = rem % 256;
  out[((long)b * 256 + o) * 225 + p] = in[((long)b * 225 + p) * 256 + o];
}

// UE[b,i,j,n] += u_emb[b,i]*sent[b,j]*Wexp[n,256]
__global__ void rank1_kernel(float* __restrict__ UE, const float* __restrict__ uemb,
                             const float* __restrict__ sentv, const float* __restrict__ Wexp) {
  long idx = (long)blockIdx.x * blockDim.x + threadIdx.x;
  if (idx >= 16777216L) return;
  int n = (int)(idx & 255);
  long t = idx >> 8;
  int j = (int)(t & 127);
  int i = (int)((t >> 7) & 127);
  int b = (int)(t >> 14);
  UE[idx] += uemb[b * 128 + i] * sentv[b * 128 + j] * Wexp[(long)n * 257 + 256];
}

// ---------------------------------------------------------------------------
// Host side
// ---------------------------------------------------------------------------
struct GB {
  int nz = 1;
  int zdA = 1; long zhA = 0, zlA = 0;
  int zdB = 1; long zhB = 0, zlB = 0;
  int zdC = 1; long zhC = 0, zlC = 0;
  int mdiv = 1 << 30; long pMhi = 0; long pMlo = -1;
  int ndiv = 1 << 30; long pNhi = 0;
};

static void gemm(hipStream_t st, int act, bool bt,
                 const float* A, const float* B, const float* bias, const float* res, float* C,
                 int M, int N, int K, int lda, int ldb, int ldc, GB g = GB()) {
  if (g.pMlo < 0) g.pMlo = ldc;
  dim3 gr((N + 63) / 64, (M + 63) / 64, g.nz);
  int sel = act * 2 + (bt ? 1 : 0);
#define GA A,B,bias,res,C,M,N,K,lda,ldb,g.zdA,g.zhA,g.zlA,g.zdB,g.zhB,g.zlB,g.zdC,g.zhC,g.zlC,g.mdiv,g.pMhi,g.pMlo,g.ndiv,g.pNhi
  switch (sel) {
    case 0: gemm_wmma<0,false><<<gr,256,0,st>>>(GA); break;
    case 1: gemm_wmma<0,true ><<<gr,256,0,st>>>(GA); break;
    case 2: gemm_wmma<1,false><<<gr,256,0,st>>>(GA); break;
    case 3: gemm_wmma<1,true ><<<gr,256,0,st>>>(GA); break;
    case 4: gemm_wmma<2,false><<<gr,256,0,st>>>(GA); break;
    case 5: gemm_wmma<2,true ><<<gr,256,0,st>>>(GA); break;
  }
#undef GA
}

struct LayerP {
  const float *Bg,*Wp,*bp,*ln_g,*ln_b,*Wv,*Win;
  const float *pxWt,*pxlng,*pxlnb,*pxW1,*pxb1,*pxW2,*pxb2;
  const float *pyWt,*pylng,*pylnb,*pyW1,*pyb1,*pyW2,*pyb2;
  const float *Wqkx,*Wqky,*gn_g,*gn_b,*Wo1,*Wo2;
};
struct Pm {
  const float *u,*grid,*sent;
  const float *Win0,*bin0,*Wc1,*bc1,*Wc2,*bc2,*Wx1,*bx1,*Wx2,*bx2,*Wx3,*bx3;
  const float *We1,*be1,*We2,*be2,*Ws1,*bs1,*Ws2,*bs2,*Ws3,*bs3,*Ws4,*bs4;
  const float *Wexp,*lt_emb,*pg,*pb,*pW1,*pb1,*pW2,*pb2,*og_g,*og_b,*Wg1,*W2c,*W3c,*b3c;
  LayerP L[2];
};

static void fill_params(void* const* d_in, const int* in_sizes, Pm& p) {
  auto F = [&](int i) { return (const float*)d_in[i]; };
  bool topA = (in_sizes[0] == 2621440);          // insertion: u first; sorted: grid first
  int pbase = topA ? 4 : 2;
  if (topA) { p.u = F(0); p.grid = F(1); p.sent = F(3); }
  else      { p.grid = F(0); p.sent = F(94); p.u = F(95); }
  bool psorted = (in_sizes[pbase] == 32768);     // sorted params: first leaf W2c (32768)
  const float* q[92];
  for (int i = 0; i < 92; ++i) q[i] = F(pbase + i);
  if (!psorted) {
    p.Win0=q[0]; p.bin0=q[1];
    for (int l = 0; l < 2; ++l) {
      const float* const* r = q + 2 + l * 27; LayerP& L = p.L[l];
      L.Bg=r[0]; L.Wp=r[1]; L.bp=r[2]; L.ln_g=r[3]; L.ln_b=r[4]; L.Wv=r[5]; L.Win=r[6];
      L.pxWt=r[7]; L.pxlng=r[8]; L.pxlnb=r[9]; L.pxW1=r[10]; L.pxb1=r[11]; L.pxW2=r[12]; L.pxb2=r[13];
      L.pyWt=r[14]; L.pylng=r[15]; L.pylnb=r[16]; L.pyW1=r[17]; L.pyb1=r[18]; L.pyW2=r[19]; L.pyb2=r[20];
      L.Wqkx=r[21]; L.Wqky=r[22]; L.gn_g=r[23]; L.gn_b=r[24]; L.Wo1=r[25]; L.Wo2=r[26];
    }
    p.Wc1=q[56]; p.bc1=q[57]; p.Wc2=q[58]; p.bc2=q[59];
    p.Wx1=q[60]; p.bx1=q[61]; p.Wx2=q[62]; p.bx2=q[63]; p.Wx3=q[64]; p.bx3=q[65];
    p.We1=q[66]; p.be1=q[67]; p.We2=q[68]; p.be2=q[69];
    p.Ws1=q[70]; p.bs1=q[71]; p.Ws2=q[72]; p.bs2=q[73]; p.Ws3=q[74]; p.bs3=q[75]; p.Ws4=q[76]; p.bs4=q[77];
    p.Wexp=q[78]; p.lt_emb=q[79]; p.pg=q[80]; p.pb=q[81]; p.pW1=q[82]; p.pb1=q[83]; p.pW2=q[84]; p.pb2=q[85];
    p.og_g=q[86]; p.og_b=q[87]; p.Wg1=q[88]; p.W2c=q[89]; p.W3c=q[90]; p.b3c=q[91];
  } else {
    p.W2c=q[0]; p.W3c=q[1]; p.Wc1=q[2]; p.Wc2=q[3]; p.We1=q[4]; p.We2=q[5]; p.Wexp=q[6]; p.Wg1=q[7];
    p.Win0=q[8]; p.Ws1=q[9]; p.Ws2=q[10]; p.Ws3=q[11]; p.Ws4=q[12]; p.Wx1=q[13]; p.Wx2=q[14]; p.Wx3=q[15];
    p.b3c=q[16]; p.bc1=q[17]; p.bc2=q[18]; p.be1=q[19]; p.be2=q[20]; p.bin0=q[21];
    p.bs1=q[22]; p.bs2=q[23]; p.bs3=q[24]; p.bs4=q[25]; p.bx1=q[26]; p.bx2=q[27]; p.bx3=q[28];
    for (int l = 0; l < 2; ++l) {
      const float* const* r = q + 29 + l * 27; LayerP& L = p.L[l];
      L.Bg=r[0]; L.Wp=r[1]; L.bp=r[2];
      L.Win=r[3]; L.Wo1=r[4]; L.Wo2=r[5]; L.Wv=r[6]; L.gn_b=r[7]; L.gn_g=r[8];
      L.Wqkx=r[9]; L.Wqky=r[10]; L.ln_b=r[11]; L.ln_g=r[12];
      L.pxW1=r[13]; L.pxW2=r[14]; L.pxWt=r[15]; L.pxb1=r[16]; L.pxb2=r[17]; L.pxlnb=r[18]; L.pxlng=r[19];
      L.pyW1=r[20]; L.pyW2=r[21]; L.pyWt=r[22]; L.pyb1=r[23]; L.pyb2=r[24]; L.pylnb=r[25]; L.pylng=r[26];
    }
    p.lt_emb=q[83]; p.og_b=q[84]; p.og_g=q[85]; p.pW1=q[86]; p.pW2=q[87]; p.pb=q[88]; p.pb1=q[89]; p.pb2=q[90]; p.pg=q[91];
  }
}

extern "C" void kernel_launch(void* const* d_in, const int* in_sizes, int n_in,
                              void* d_out, int out_size, void* d_ws, size_t ws_size,
                              hipStream_t stream) {
  (void)n_in; (void)out_size; (void)ws_size;
  Pm p; fill_params(d_in, in_sizes, p);
  hipStream_t st = stream;

  float* w = (float*)d_ws;
  size_t off = 0;
  auto alloc = [&](size_t n) { float* r = w + off; off += (n + 63) & ~(size_t)63; return r; };
  float* U     = alloc(16777216);  // tokens (b,nx,ny,256)
  float* UN    = alloc(16777216);
  float* UI    = alloc(16777216);
  float* SLABA = alloc(16777216);  // V (per-b) / latent m1
  float* SLABB = alloc(8388608);   // O1 / GO1
  float* SLABC = alloc(8388608);   // O2
  float* SLABD = alloc(8388608);   // OG / final T2
  float* S     = alloc(4194304);
  float* PE    = alloc(4194304);
  float* QK    = alloc(1048576);
  float* QR    = alloc(524288);
  float* KR    = alloc(524288);
  float* KX    = alloc(524288);
  float* KY    = alloc(524288);
  float* UIMX  = alloc(131072);
  float* UIMY  = alloc(131072);
  float* T0    = alloc(131072);
  float* T1s   = alloc(131072);
  float* UX    = alloc(131072);
  float* UY    = alloc(131072);
  float* POS   = alloc(256);
  float* STAT  = alloc(64);
  float* CONV1 = alloc(230400);
  float* CONV2 = alloc(230400);
  float* XC    = alloc(230400);
  float* H1    = alloc(262144);
  float* H2    = alloc(262144);
  float* HVb   = alloc(1024);
  float* UE1b  = alloc(1024);
  float* UEMB  = alloc(512);
  float* S1b   = alloc(2048);
  float* S2b   = alloc(2048);
  float* S3b   = alloc(1024);
  float* SENTV = alloc(512);

  const float eps = 1e-5f;
  // ---- input projection: u = u @ Win0^T + bin0
  gemm(st, 0, true, p.u, p.Win0, p.bin0, nullptr, U, 65536, 256, 40, 40, 40, 256);
  extract_pos_kernel<<<1, 128, 0, st>>>(p.grid, POS);

  for (int l = 0; l < 2; ++l) {
    const LayerP& L = p.L[l];
    // positional encoding
    pe_feat_kernel<<<(2097152 + 255) / 256, 256, 0, st>>>(POS, POS + 128, L.Bg, S);
    gemm(st, 0, true, S, L.Wp, L.bp, nullptr, PE, 16384, 256, 256, 256, 256, 256);
    add_bcast_kernel<<<(16777216 + 255) / 256, 256, 0, st>>>(U, PE, 16777216L, 4194304L);
    // LN + projections
    ln_kernel<<<8192, 256, 0, st>>>(U, L.ln_g, L.ln_b, UN, 65536, 256, eps);
    gemm(st, 0, true, UN, L.Win, nullptr, nullptr, UI, 65536, 256, 256, 256, 256, 256);
    // pool means (mean commutes with the linear Wt)
    pool_mean_kernel<<<512, 256, 0, st>>>(UI, UIMX, 1, 32768L, 0L, 256L, 128);
    pool_mean_kernel<<<512, 256, 0, st>>>(UI, UIMY, 128, 4194304L, 256L, 32768L, 128);
    // px chain -> UX
    gemm(st, 0, true, UIMX, L.pxWt, nullptr, nullptr, T0, 512, 256, 256, 256, 256, 256);
    ln_kernel<<<64, 256, 0, st>>>(T0, L.pxlng, L.pxlnb, T1s, 512, 256, eps);
    gemm(st, 1, true, T1s, L.pxW1, L.pxb1, nullptr, T0, 512, 256, 256, 256, 256, 256);
    gemm(st, 0, true, T0, L.pxW2, L.pxb2, nullptr, UX, 512, 256, 256, 256, 256, 256);
    // py chain -> UY
    gemm(st, 0, true, UIMY, L.pyWt, nullptr, nullptr, T0, 512, 256, 256, 256, 256, 256);
    ln_kernel<<<64, 256, 0, st>>>(T0, L.pylng, L.pylnb, T1s, 512, 256, eps);
    gemm(st, 1, true, T1s, L.pyW1, L.pyb1, nullptr, T0, 512, 256, 256, 256, 256, 256);
    gemm(st, 0, true, T0, L.pyW2, L.pyb2, nullptr, UY, 512, 256, 256, 256, 256, 256);
    // kx = rope(q) @ rope(k)^T, batched over (b,h)
    gemm(st, 0, true, UX, L.Wqkx, nullptr, nullptr, QK, 512, 2048, 256, 256, 256, 2048);
    rope_kernel<<<(524288 + 255) / 256, 256, 0, st>>>(QK, POS, QR, KR);
    { GB g; g.nz = 32; g.zhA = 16384; g.zhB = 16384; g.zhC = 16384;
      gemm(st, 0, true, QR, KR, nullptr, nullptr, KX, 128, 128, 128, 128, 128, 128, g); }
    gemm(st, 0, true, UY, L.Wqky, nullptr, nullptr, QK, 512, 2048, 256, 256, 256, 2048);
    rope_kernel<<<(524288 + 255) / 256, 256, 0, st>>>(QK, POS + 128, QR, KR);
    { GB g; g.nz = 32; g.zhA = 16384; g.zhB = 16384; g.zhC = 16384;
      gemm(st, 0, true, QR, KR, nullptr, nullptr, KY, 128, 128, 128, 128, 128, 128, g); }
    // attention body per batch
    for (int b = 0; b < 4; ++b) {
      const float* UNb = UN + (long)b * 4194304;
      float* Ub = U + (long)b * 4194304;
      // V = (un @ Wv^T) stored permuted as (h, nx, ny, c)
      { GB g; g.pMlo = 64; g.ndiv = 64; g.pNhi = 1048576;
        gemm(st, 0, true, UNb, L.Wv, nullptr, nullptr, SLABA, 16384, 512, 256, 256, 256, 0, g); }
      // O1[h](i, l*c) = KX[h](i,j) @ V[h](j, l*c)
      { GB g; g.nz = 8; g.zhA = 16384; g.zhB = 1048576; g.zhC = 1048576; g.pMlo = 8192;
        gemm(st, 0, false, KX + (long)b * 131072, SLABA, nullptr, nullptr, SLABB,
             128, 8192, 128, 128, 8192, 0, g); }
      // O2[h,i](l,c) = KY[h](l,m) @ O1[h,i](m,c)
      { GB g; g.nz = 1024;
        g.zdA = 128; g.zhA = 16384; g.zlA = 0;
        g.zdB = 128; g.zhB = 1048576; g.zlB = 8192;
        g.zdC = 128; g.zhC = 1048576; g.zlC = 8192; g.pMlo = 64;
        gemm(st, 0, false, KY + (long)b * 131072, SLABB, nullptr, nullptr, SLABC,
             128, 64, 128, 128, 64, 0, g); }
      // GroupNorm over each head (contiguous 1M floats per head)
      gn_stats_kernel<<<8, 256, 0, st>>>(SLABC, STAT, 1, 1048576L, 0L, 1048576, 1, 1L, eps);
      og_apply_attn_kernel<<<(8388608 + 255) / 256, 256, 0, st>>>(SLABC, STAT, L.gn_g, L.gn_b, SLABD);
      // out proj + residual
      gemm(st, 1, true, SLABD, L.Wo1, nullptr, nullptr, SLABB, 16384, 256, 512, 512, 512, 256);
      gemm(st, 0, true, SLABB, L.Wo2, nullptr, Ub, Ub, 16384, 256, 256, 256, 256, 256);
    }
  }

  // ---- conv head -> u_emb
  conv_gemm_kernel<<<dim3(4, 15, 1), 256, 0, st>>>(U, p.Wc1, p.bc1, CONV1);
  gemm(st, 0, true, CONV1, p.Wc2, p.bc2, nullptr, CONV2, 900, 256, 256, 256, 256, 256);
  transpose_conv_kernel<<<(230400 + 255) / 256, 256, 0, st>>>(CONV2, XC);
  gemm(st, 2, true, XC, p.Wx1, p.bx1, nullptr, H1, 1024, 256, 225, 225, 225, 256);
  gemm(st, 2, true, H1, p.Wx2, p.bx2, nullptr, H2, 1024, 256, 256, 256, 256, 256);
  gemm(st, 0, true, H2, p.Wx3, p.bx3, nullptr, HVb, 1024, 1, 256, 256, 256, 1);
  gemm(st, 2, true, HVb, p.We1, p.be1, nullptr, UE1b, 4, 256, 256, 256, 256, 256);
  gemm(st, 0, true, UE1b, p.We2, p.be2, nullptr, UEMB, 4, 128, 256, 256, 256, 128);
  // ---- sentence head
  gemm(st, 2, true, p.sent, p.Ws1, p.bs1, nullptr, S1b, 4, 512, 768, 768, 768, 512);
  gemm(st, 2, true, S1b, p.Ws2, p.bs2, nullptr, S2b, 4, 512, 512, 512, 512, 512);
  gemm(st, 2, true, S2b, p.Ws3, p.bs3, nullptr, S3b, 4, 256, 512, 512, 512, 256);
  gemm(st, 0, true, S3b, p.Ws4, p.bs4, nullptr, SENTV, 4, 128, 256, 256, 256, 128);
  // ---- Wexp (first 256 cols) + rank-1 term from concatenated emb channel
  gemm(st, 0, true, U, p.Wexp, nullptr, nullptr, UN, 65536, 256, 256, 256, 257, 256);
  rank1_kernel<<<(16777216 + 255) / 256, 256, 0, st>>>(UN, UEMB, SENTV, p.Wexp);
  // ---- latent step 0 (only step 0 reaches the output)
  add_bcast_kernel<<<(16777216 + 255) / 256, 256, 0, st>>>(UN, p.lt_emb, 16777216L, 256L);
  ln_kernel<<<8192, 256, 0, st>>>(UN, p.pg, p.pb, UI, 65536, 256, eps);
  gemm(st, 1, true, UI, p.pW1, p.pb1, nullptr, SLABA, 65536, 256, 256, 256, 256, 256);
  gemm(st, 0, true, SLABA, p.pW2, p.pb2, UN, U, 65536, 256, 256, 256, 256, 256);
  // ---- output head: GN(8 groups) -> block-diag Wg1 (GELU) -> W2c (GELU) -> W3c
  gn_stats_kernel<<<32, 256, 0, st>>>(U, STAT, 8, 4194304L, 32L, 16384, 32, 256L, eps);
  og_apply_final_kernel<<<(16777216 + 255) / 256, 256, 0, st>>>(U, STAT, p.og_g, p.og_b, UN);
  { GB g; g.nz = 32;
    g.zdA = 8; g.zhA = 4194304; g.zlA = 32;
    g.zdB = 8; g.zhB = 0;       g.zlB = 1024;
    g.zdC = 8; g.zhC = 4194304; g.zlC = 32; g.pMlo = 256;
    gemm(st, 1, true, UN, p.Wg1, nullptr, nullptr, UI, 16384, 32, 32, 256, 32, 0, g); }
  gemm(st, 1, true, UI, p.W2c, nullptr, nullptr, SLABD, 65536, 128, 256, 256, 256, 128);
  gemm(st, 0, true, SLABD, p.W3c, p.b3c, nullptr, (float*)d_out, 65536, 4, 128, 128, 128, 4);
}
